// MultiHeadedAttention_16183436772160
// MI455X (gfx1250) — matrix operand order repaired and checked by reading, not compiled, not run on hardware
//
#include <hip/hip_runtime.h>

// ---------------- constants ----------------
static constexpr int kB  = 16;    // batch
static constexpr int kH  = 16;    // heads
static constexpr int kS  = 768;   // sequence (n + nk)
static constexpr int kD  = 1024;  // d_model
static constexpr int kDk = 64;    // head dim
static constexpr int kN  = 512;   // code length n
static constexpr int kNk = 256;   // redundancy n-k
static constexpr int kM  = kB * kS; // 12288 GEMM rows

typedef __attribute__((ext_vector_type(16))) _Float16 v16h;
typedef __attribute__((ext_vector_type(8)))  _Float16 v8h;
typedef __attribute__((ext_vector_type(4)))  _Float16 v4h;
typedef __attribute__((ext_vector_type(8)))  float    v8f;
typedef __attribute__((ext_vector_type(4)))  float    v4f;

static __device__ __forceinline__ v16h comb16(v8h a, v8h b) {
  return __builtin_shufflevector(a, b, 0,1,2,3,4,5,6,7,8,9,10,11,12,13,14,15);
}
static __device__ __forceinline__ v8f wmma_f16(v16h a, v16h b, v8f c) {
  // D = A(16x32 f16) * B(32x16 f16) + C(16x16 f32)
  return __builtin_amdgcn_wmma_f32_16x16x32_f16(false, a, false, b, (short)0, c,
                                                false, false);
}

// ---- 16-lane butterfly reductions in pure VALU via DPP16 (no LDS traffic) ----
// masks {1,2,7,15}: quad_perm(1,0,3,2)=0xB1, quad_perm(2,3,0,1)=0x4E,
// row_half_mirror(l^7)=0x141, row_mirror(l^15)=0x140; span = full 16-lane group.
template <int CTRL>
static __device__ __forceinline__ float dppf(float x) {
  int v = __float_as_int(x);
  v = __builtin_amdgcn_update_dpp(v, v, CTRL, 0xF, 0xF, true);
  return __int_as_float(v);
}
static __device__ __forceinline__ float rmax16(float x) {
  x = fmaxf(x, dppf<0xB1>(x));
  x = fmaxf(x, dppf<0x4E>(x));
  x = fmaxf(x, dppf<0x141>(x));
  x = fmaxf(x, dppf<0x140>(x));
  return x;
}
static __device__ __forceinline__ float rsum16(float x) {
  x += dppf<0xB1>(x);
  x += dppf<0x4E>(x);
  x += dppf<0x141>(x);
  x += dppf<0x140>(x);
  return x;
}

// ---------------- mask construction ----------------
// nn[i][j] = sum_k pc[k][i] * pc[k][j]   (512x512, k over 256)
__global__ void pc_nn_kernel(const float* __restrict__ pc, float* __restrict__ nn) {
  int j = blockIdx.x * blockDim.x + threadIdx.x; // 0..511
  int i = blockIdx.y;                            // 0..511
  float acc = 0.f;
  for (int k = 0; k < kNk; ++k)
    acc += pc[(size_t)k * kN + i] * pc[(size_t)k * kN + j];
  nn[(size_t)i * kN + j] = acc;
}

// nknk[i][j] = sum_k pc[i][k] * pc[j][k]  (256x256, k over 512)
__global__ void pc_nknk_kernel(const float* __restrict__ pc, float* __restrict__ nknk) {
  int j = blockIdx.x * blockDim.x + threadIdx.x; // 0..255
  int i = blockIdx.y;                            // 0..255
  float acc = 0.f;
  for (int k = 0; k < kN; ++k)
    acc += pc[(size_t)i * kN + k] * pc[(size_t)j * kN + k];
  nknk[(size_t)i * kNk + j] = acc;
}

// mask[i][j] = MLP(M[i][j]) with M the block matrix [[nn, pc^T],[pc, nknk]]
__global__ void mask_mlp_kernel(const float* __restrict__ pc,
                                const float* __restrict__ nn,
                                const float* __restrict__ nknk,
                                const float* __restrict__ mw1,
                                const float* __restrict__ mb1,
                                const float* __restrict__ mw2,
                                const float* __restrict__ mb2,
                                float* __restrict__ mask) {
  __shared__ float w1[50], b1[50], w2[50];
  int t = threadIdx.x;
  if (t < 50) { w1[t] = mw1[t]; b1[t] = mb1[t]; w2[t] = mw2[t]; }
  __syncthreads();
  int idx = blockIdx.x * blockDim.x + t;
  if (idx >= kS * kS) return;
  int i = idx / kS, j = idx % kS;
  float m;
  if (i < kN) m = (j < kN) ? nn[(size_t)i * kN + j] : pc[(size_t)(j - kN) * kN + i];
  else        m = (j < kN) ? pc[(size_t)(i - kN) * kN + j]
                           : nknk[(size_t)(i - kN) * kNk + (j - kN)];
  float acc = mb2[0];
  #pragma unroll
  for (int hidx = 0; hidx < 50; ++hidx) {
    float v = m * w1[hidx] + b1[hidx];
    v = v > 0.f ? v : 0.f;
    acc += v * w2[hidx];
  }
  mask[idx] = acc;
}

// ---------------- WMMA GEMM: C = A(12288x1024) @ W(1024x1024) + bias ----------------
// A_HALF: A is f16 (else fp32, converted while staging to LDS).
// MODE 0: write f16 scaled output to (b,h,s,d) layout. MODE 1: write fp32 (B,S,D).
template <bool A_HALF, int MODE>
__global__ __launch_bounds__(256)
void gemm_kernel(const void* __restrict__ Aptr, const float* __restrict__ W,
                 const float* __restrict__ bias, void* __restrict__ Outp, float scale) {
  __shared__ alignas(16) _Float16 As[128 * 40];   // 128 rows x 32 k (pad 40)
  __shared__ alignas(16) _Float16 BsT[128 * 40];  // 128 n x 32 k (pad 40), transposed

  const int t = threadIdx.x;
  const int wave = t >> 5, lane = t & 31, lr = lane & 15, lh = lane >> 4;
  const int wm = wave >> 2, wn = wave & 3;        // waves 2(M) x 4(N)
  const int m0 = blockIdx.y * 128, n0 = blockIdx.x * 128;

  v8f zero = {};
  v8f acc[4][2];
  #pragma unroll
  for (int i = 0; i < 4; ++i)
    #pragma unroll
    for (int j = 0; j < 2; ++j) acc[i][j] = zero;

  for (int kb = 0; kb < kD; kb += 32) {
    __syncthreads();
    if (A_HALF) {
      // stage A: 16B vector copies (2 per thread)
      #pragma unroll
      for (int l = t; l < 128 * 4; l += 256) {
        int r = l >> 2, c8 = (l & 3) * 8;
        *(v8h*)(As + r * 40 + c8) =
            *(const v8h*)((const _Float16*)Aptr + (size_t)(m0 + r) * kD + kb + c8);
      }
    } else {
      // stage A: float4 load -> packed half4 store (4 per thread)
      #pragma unroll
      for (int l = t; l < 128 * 8; l += 256) {
        int r = l >> 3, c4 = (l & 7) * 4;
        v4f f = *(const v4f*)((const float*)Aptr + (size_t)(m0 + r) * kD + kb + c4);
        v4h hv;
        hv[0] = (_Float16)f[0]; hv[1] = (_Float16)f[1];
        hv[2] = (_Float16)f[2]; hv[3] = (_Float16)f[3];
        *(v4h*)(As + r * 40 + c4) = hv;
      }
    }
    // stage W transposed: BsT[n][k]; 4 strided loads -> one 8B LDS store
    #pragma unroll
    for (int l = t; l < 1024; l += 256) {
      int n = l & 127, r4 = (l >> 7) * 4;
      const float* wp = W + (size_t)(kb + r4) * kD + n0 + n;
      v4h hv;
      hv[0] = (_Float16)wp[0];
      hv[1] = (_Float16)wp[kD];
      hv[2] = (_Float16)wp[2 * kD];
      hv[3] = (_Float16)wp[3 * kD];
      *(v4h*)(BsT + n * 40 + r4) = hv;
    }
    __syncthreads();

    v16h af[4], bf[2];
    #pragma unroll
    for (int mt = 0; mt < 4; ++mt) {
      const _Float16* p = As + (wm * 64 + mt * 16 + lr) * 40;
      af[mt] = comb16(*(const v8h*)(p + lh * 8), *(const v8h*)(p + 16 + lh * 8));
    }
    #pragma unroll
    for (int nt = 0; nt < 2; ++nt) {
      const _Float16* p = BsT + (wn * 32 + nt * 16 + lr) * 40;
      bf[nt] = comb16(*(const v8h*)(p + lh * 8), *(const v8h*)(p + 16 + lh * 8));
    }
    #pragma unroll
    for (int mt = 0; mt < 4; ++mt)
      #pragma unroll
      for (int nt = 0; nt < 2; ++nt)
        acc[mt][nt] = wmma_f16(af[mt], bf[nt], acc[mt][nt]);
  }

  #pragma unroll
  for (int mt = 0; mt < 4; ++mt) {
    #pragma unroll
    for (int nt = 0; nt < 2; ++nt) {
      const int col = n0 + wn * 32 + nt * 16 + lr;
      const float bv = bias[col];
      #pragma unroll
      for (int v = 0; v < 8; ++v) {
        const int row = m0 + wm * 64 + mt * 16 + v + 8 * lh; // C layout: M = v + 8*(lane>=16)
        const float val = (acc[mt][nt][v] + bv) * scale;
        if (MODE == 0) {
          const int bi = row / kS, s = row % kS;
          const int hh = col >> 6, d = col & 63;
          ((_Float16*)Outp)[(((size_t)bi * kH + hh) * kS + s) * kDk + d] = (_Float16)val;
        } else {
          ((float*)Outp)[(size_t)row * kD + col] = val;
        }
      }
    }
  }
}

// ---------------- flash attention with multiplicative mask ----------------
// grid: (S/128, B*H). 8 waves; wave = one 16-row q tile. K/V chunks of 32 keys in LDS.
__global__ __launch_bounds__(256)
void attention_kernel(const _Float16* __restrict__ q16, const _Float16* __restrict__ k16,
                      const _Float16* __restrict__ v16, const float* __restrict__ mask,
                      _Float16* __restrict__ x16) {
  __shared__ alignas(16) _Float16 Ks[32 * 72];      // keys x d (pad 72)
  __shared__ alignas(16) _Float16 VsT[64 * 40];     // d x keys (pad 40)
  __shared__ alignas(16) _Float16 Ps[8][16 * 40];   // per-wave P tile 16x32 (pad 40)

  const int t = threadIdx.x;
  const int wave = t >> 5, lane = t & 31, lr = lane & 15, lh = lane >> 4;
  const int bh = blockIdx.y;
  const int b = bh >> 4, h = bh & 15;
  const size_t hoff = (size_t)bh * kS * kDk;
  const int qs = blockIdx.x * 128 + wave * 16;

  // q fragments (q pre-scaled by 1/sqrt(dk)); A layout: row=lr, K chunks by lh
  const _Float16* qrow = q16 + hoff + (size_t)(qs + lr) * kDk;
  const v16h aq0 = comb16(*(const v8h*)(qrow + lh * 8),      *(const v8h*)(qrow + 16 + lh * 8));
  const v16h aq1 = comb16(*(const v8h*)(qrow + 32 + lh * 8), *(const v8h*)(qrow + 48 + lh * 8));

  float mrow[8], lrow[8];
  v8f zero = {};
  v8f ov[4];
  #pragma unroll
  for (int v = 0; v < 8; ++v) { mrow[v] = -1e30f; lrow[v] = 0.f; }
  #pragma unroll
  for (int i = 0; i < 4; ++i) ov[i] = zero;

  // staging coordinates: one 16B vector per thread covers the 32x64 chunk
  const int str = t >> 3, stc = (t & 7) * 8;

  for (int kb = 0; kb < kS; kb += 32) {
    __syncthreads(); // previous-iteration LDS readers done
    *(v8h*)(Ks + str * 72 + stc) =
        *(const v8h*)(k16 + hoff + (size_t)(kb + str) * kDk + stc);
    {
      v8h vv = *(const v8h*)(v16 + hoff + (size_t)(kb + str) * kDk + stc);
      #pragma unroll
      for (int i = 0; i < 8; ++i) VsT[(stc + i) * 40 + str] = vv[i];
    }
    __syncthreads();

    // scores for two 16-key tiles (dk=64 -> two K=32 WMMAs each)
    v8f s0 = zero, s1 = zero;
    {
      const _Float16* kp0 = Ks + lr * 72;
      const _Float16* kp1 = Ks + (16 + lr) * 72;
      v16h bk;
      bk = comb16(*(const v8h*)(kp0 + lh * 8),      *(const v8h*)(kp0 + 16 + lh * 8));
      s0 = wmma_f16(aq0, bk, s0);
      bk = comb16(*(const v8h*)(kp0 + 32 + lh * 8), *(const v8h*)(kp0 + 48 + lh * 8));
      s0 = wmma_f16(aq1, bk, s0);
      bk = comb16(*(const v8h*)(kp1 + lh * 8),      *(const v8h*)(kp1 + 16 + lh * 8));
      s1 = wmma_f16(aq0, bk, s1);
      bk = comb16(*(const v8h*)(kp1 + 32 + lh * 8), *(const v8h*)(kp1 + 48 + lh * 8));
      s1 = wmma_f16(aq1, bk, s1);
    }

    // multiplicative mask + online softmax; row r = v + 8*lh lives on 16 lanes (col = lr)
    #pragma unroll
    for (int v = 0; v < 8; ++v) {
      const int rowg = qs + v + 8 * lh;
      const float* mrp = mask + (size_t)rowg * kS + kb;
      float e0 = s0[v] * mrp[lr];
      float e1 = s1[v] * mrp[16 + lr];
      const float tmax = rmax16(fmaxf(e0, e1));
      const float newm = fmaxf(mrow[v], tmax);
      const float p0 = __expf(e0 - newm);
      const float p1 = __expf(e1 - newm);
      const float tsum = rsum16(p0 + p1);
      const float alpha = __expf(mrow[v] - newm);
      lrow[v] = lrow[v] * alpha + tsum;
      mrow[v] = newm;
      #pragma unroll
      for (int i = 0; i < 4; ++i) ov[i][v] *= alpha;
      _Float16* pr = &Ps[wave][(v + 8 * lh) * 40];
      pr[lr] = (_Float16)p0;
      pr[16 + lr] = (_Float16)p1;
    }
    __syncthreads(); // make P (cross-lane) visible + VsT still valid

    // PV: o(16x64) += P(16x32) @ V(32x64), 4 N-tiles of 16
    {
      const _Float16* pp = &Ps[wave][lr * 40];
      const v16h ap = comb16(*(const v8h*)(pp + lh * 8), *(const v8h*)(pp + 16 + lh * 8));
      #pragma unroll
      for (int dt = 0; dt < 4; ++dt) {
        const _Float16* vp = VsT + (dt * 16 + lr) * 40;
        const v16h bvf = comb16(*(const v8h*)(vp + lh * 8), *(const v8h*)(vp + 16 + lh * 8));
        ov[dt] = wmma_f16(ap, bvf, ov[dt]);
      }
    }
  }

  // normalize and store f16 in (B,S,D) order for the Wo GEMM
  #pragma unroll
  for (int v = 0; v < 8; ++v) {
    const float inv = 1.0f / lrow[v];
    const int srow = qs + v + 8 * lh;
    #pragma unroll
    for (int dt = 0; dt < 4; ++dt) {
      const int dcol = h * kDk + dt * 16 + lr;
      x16[((size_t)b * kS + srow) * kD + dcol] = (_Float16)(ov[dt][v] * inv);
    }
  }
}

// ---------------- host launch ----------------
extern "C" void kernel_launch(void* const* d_in, const int* in_sizes, int n_in,
                              void* d_out, int out_size, void* d_ws, size_t ws_size,
                              hipStream_t stream) {
  (void)in_sizes; (void)n_in; (void)out_size;
  const float* query = (const float*)d_in[0];
  const float* key_i = (const float*)d_in[1];
  const float* value = (const float*)d_in[2];
  const float* pc    = (const float*)d_in[3];
  const float* Wq = (const float*)d_in[4];  const float* bq = (const float*)d_in[5];
  const float* Wk = (const float*)d_in[6];  const float* bk = (const float*)d_in[7];
  const float* Wv = (const float*)d_in[8];  const float* bv = (const float*)d_in[9];
  const float* Wo = (const float*)d_in[10]; const float* bo = (const float*)d_in[11];
  const float* mw1 = (const float*)d_in[12]; const float* mb1 = (const float*)d_in[13];
  const float* mw2 = (const float*)d_in[14]; const float* mb2 = (const float*)d_in[15];

  const size_t QKV_HALVES = (size_t)kB * kH * kS * kDk; // 12,582,912 halves
  char* w = (char*)d_ws;
  _Float16* q16 = (_Float16*)w; w += QKV_HALVES * 2;
  _Float16* k16 = (_Float16*)w; w += QKV_HALVES * 2;
  _Float16* v16 = (_Float16*)w; w += QKV_HALVES * 2;
  _Float16* x16 = (_Float16*)w; w += QKV_HALVES * 2;
  float* mask = (float*)w; w += (size_t)kS * kS * 4;
  float* nn   = (float*)w; w += (size_t)kN * kN * 4;
  float* nknk = (float*)w; w += (size_t)kNk * kNk * 4;
  if ((size_t)(w - (char*)d_ws) > ws_size) return; // ~100 MB scratch required

  // 1) mask
  pc_nn_kernel<<<dim3(2, 512), 256, 0, stream>>>(pc, nn);
  pc_nknk_kernel<<<dim3(1, 256), 256, 0, stream>>>(pc, nknk);
  mask_mlp_kernel<<<(kS * kS) / 256, 256, 0, stream>>>(pc, nn, nknk, mw1, mb1, mw2, mb2, mask);

  // 2) projections (q scaled by 1/sqrt(dk)); f16 output in (b,h,s,d)
  gemm_kernel<false, 0><<<dim3(kD / 128, kM / 128), 256, 0, stream>>>(query, Wq, bq, q16, 0.125f);
  gemm_kernel<false, 0><<<dim3(kD / 128, kM / 128), 256, 0, stream>>>(key_i, Wk, bk, k16, 1.0f);
  gemm_kernel<false, 0><<<dim3(kD / 128, kM / 128), 256, 0, stream>>>(value, Wv, bv, v16, 1.0f);

  // 3) masked attention, flash-style
  attention_kernel<<<dim3(kS / 128, kB * kH), 256, 0, stream>>>(q16, k16, v16, mask, x16);

  // 4) output projection -> fp32 d_out
  gemm_kernel<true, 1><<<dim3(kD / 128, kM / 128), 256, 0, stream>>>(x16, Wo, bo, d_out, 1.0f);
}
